// ZipformerBlock_16028817949126
// MI455X (gfx1250) — compile-verified, matrix operand order checked by reading, not run on hardware
//
#include <hip/hip_runtime.h>
#include <hip/hip_bf16.h>
#include <cstdint>

// ---------------------------------------------------------------------------
// Zipformer block for MI455X (gfx1250, wave32).
//  * every matmul:   v_wmma_f32_16x16x32_f16
//  * LDS staging:    TENSOR_LOAD_TO_LDS (Tensor Data Mover, TENSORcnt),
//                    double-buffered; TDM pad fields produce the bank-skewed
//                    LDS row layout; TDM tensor-dim clamp zero-fills N edges.
//  * every GEMM B operand normalized to f16 [N,K] row-major (weights
//    transpose+cast once; activation B operands produced pre-transposed)
// ---------------------------------------------------------------------------

typedef __attribute__((ext_vector_type(16))) _Float16 v16h;
typedef __attribute__((ext_vector_type(8)))  float    v8f;
typedef __attribute__((ext_vector_type(8)))  _Float16 h8;
typedef __attribute__((ext_vector_type(4)))  uint32_t u32x4;
typedef __attribute__((ext_vector_type(8)))  uint32_t u32x8;

#define DIMC     512
#define HEADS    8
#define DIM_HEAD 64
#define NSEQ     1024
#define BATCH    8
#define TOKENS   (BATCH*NSEQ)     /* 8192 */
#define FFH      2048
#define D34      384
#define DH34     48
#define INNER2   2048
#define KW       31
#define MAXPOS   512
#define SREL     1056             /* 1025 padded */

// ---------------------------- device helpers -------------------------------
__device__ __forceinline__ float softplusf(float x) {
  return x > 20.f ? x : log1pf(__expf(x));
}
__device__ __forceinline__ float act_apply(float x, int act) {
  if (act == 1) return softplusf(x - 4.f) - 0.08f * x - 0.035f;            // swooshL
  if (act == 2) return softplusf(x - 1.f) - 0.08f * x - 0.313261687f;      // swooshR
  if (act == 3) return tanhf(x);
  return x;
}

// ------------------------------ WMMA GEMM ----------------------------------
// C[M,N] = act(alpha * A x B^T + bias) (+res). A: f16 [M,K]; B: f16 [N,K].
// Batched: z -> (outer = z/inner, inner = z%inner) with per-operand strides.
struct GemmP {
  const _Float16* A; long long aso, asi; int lda;
  const _Float16* B; long long bso, bsi; int ldb;
  float* o32; _Float16* o16; _Float16* o16t;
  long long cso, csi; int ldc, ldct;
  const float* bias; const float* res;
  int M, N, K, inner; float alpha; int act;
};

#define LDSP 40   // padded row stride in halfs (80B: 16B-aligned, bank-skewed)

// Build D# group1 for a [rows x 32half] tile copy with stride `ld` (halfs):
//   data_size=1 (2B), pad_enable, pad_interval=3 (16 DW), pad_amount=3 (4 DW)
//   tensor_dim0=32 (K chunk always full), tensor_dim1=remain (edge zero-fill)
__device__ __forceinline__ u32x8 tdm_g1(int rows, int remain, int ld) {
  u32x8 d;
  d[0] = (1u << 16) | (1u << 20) | (3u << 22) | (3u << 25);
  d[1] = 32u << 16;                                   // tensor_dim0[15:0]
  d[2] = ((uint32_t)remain & 0xFFFFu) << 16;          // dim0 hi=0 | tensor_dim1 lo
  d[3] = ((uint32_t)remain >> 16) | (32u << 16);      // tensor_dim1 hi | tile_dim0
  d[4] = (uint32_t)rows;                              // tile_dim1 | tile_dim2=0
  d[5] = (uint32_t)ld;                                // tensor_dim0_stride lo32
  d[6] = 0u;                                          // stride hi | dim1_stride lo
  d[7] = 0u;
  return d;
}
__device__ __forceinline__ void tdm_issue(uint32_t lds, const _Float16* gp, u32x8 g1) {
  const uint64_t ga = (uint64_t)(uintptr_t)gp;
  u32x4 g0;
  g0[0] = 1u;                                                    // count=1
  g0[1] = lds;                                                   // lds_addr
  g0[2] = (uint32_t)ga;                                          // global_addr lo
  g0[3] = ((uint32_t)(ga >> 32) & 0x01FFFFFFu) | (2u << 30);     // addr hi | type=2
  asm volatile("tensor_load_to_lds %0, %1" :: "s"(g0), "s"(g1) : "memory");
}

template <int TN>
__launch_bounds__(256)
__global__ void gemm_wmma(GemmP g) {
  constexpr int FR = TN / 32;             // C frags per wave (2 or 4)
  __shared__ _Float16 As[2][64 * LDSP];   // A tile [m][k]
  __shared__ _Float16 Bs[2][TN * LDSP];   // B tile [n][k]

  const int tid  = threadIdx.x;
  const int lane = tid & 31;
  const int w    = tid >> 5;
  const int wr   = w >> 1;                // 0..3 -> 16-row strip
  const int wc   = w & 1;                 // 0..1 -> TN/2-col strip
  const int zb   = blockIdx.z;
  const long long ao = (long long)(zb / g.inner) * g.aso + (long long)(zb % g.inner) * g.asi;
  const long long bo = (long long)(zb / g.inner) * g.bso + (long long)(zb % g.inner) * g.bsi;
  const long long co = (long long)(zb / g.inner) * g.cso + (long long)(zb % g.inner) * g.csi;
  const _Float16* Ag = g.A + ao;
  const _Float16* Bg = g.B + bo;
  const int m0 = blockIdx.y * 64;
  const int n0 = blockIdx.x * TN;

  const uint32_t ldsA[2] = {(uint32_t)(uintptr_t)&As[0][0], (uint32_t)(uintptr_t)&As[1][0]};
  const uint32_t ldsB[2] = {(uint32_t)(uintptr_t)&Bs[0][0], (uint32_t)(uintptr_t)&Bs[1][0]};
  const u32x8 g1A = tdm_g1(64, 64, g.lda);
  const u32x8 g1B = tdm_g1(TN, g.N - n0, g.ldb);

  v8f acc[FR];
#pragma unroll
  for (int f = 0; f < FR; f++) acc[f] = (v8f){};

  auto compute = [&](int buf) {
    // A frag (ISA 7.12.2): lane<16 -> row=lane, K=0..7 & 16..23; lane>=16 -> +8
    const int kb   = (lane >> 4) * 8;
    const int arow = wr * 16 + (lane & 15);
    h8 alo = *(const h8*)&As[buf][arow * LDSP + kb];
    h8 ahi = *(const h8*)&As[buf][arow * LDSP + kb + 16];
    v16h a;
#pragma unroll
    for (int i = 0; i < 8; i++) { a[i] = alo[i]; a[8 + i] = ahi[i]; }
#pragma unroll
    for (int f = 0; f < FR; f++) {
      const int bc = wc * (TN / 2) + f * 16 + (lane & 15);
      h8 blo = *(const h8*)&Bs[buf][bc * LDSP + kb];
      h8 bhi = *(const h8*)&Bs[buf][bc * LDSP + kb + 16];
      v16h b;
#pragma unroll
      for (int i = 0; i < 8; i++) { b[i] = blo[i]; b[8 + i] = bhi[i]; }
      acc[f] = __builtin_amdgcn_wmma_f32_16x16x32_f16(false, a, false, b,
                                                      (short)0, acc[f], false, false);
    }
  };

  // wave 0 drives the Tensor Data Mover; waves 1..7 only compute
  if (w == 0) {
    tdm_issue(ldsA[0], Ag + (long long)m0 * g.lda, g1A);
    tdm_issue(ldsB[0], Bg + (long long)n0 * g.ldb, g1B);
  }
  int k0 = 0;
  for (; k0 + 32 < g.K; k0 += 32) {                 // steady state: branch-free wait
    const int buf = (k0 >> 5) & 1;
    if (w == 0) {
      tdm_issue(ldsA[buf ^ 1], Ag + (long long)m0 * g.lda + (k0 + 32), g1A);
      tdm_issue(ldsB[buf ^ 1], Bg + (long long)n0 * g.ldb + (k0 + 32), g1B);
      __builtin_amdgcn_s_wait_tensorcnt(2);         // current stage landed
    }
    __syncthreads();
    compute(buf);
    __syncthreads();
  }
  {                                                 // peeled last K-step
    const int buf = (k0 >> 5) & 1;
    if (w == 0) __builtin_amdgcn_s_wait_tensorcnt(0);
    __syncthreads();
    compute(buf);
  }

  // ---- fused epilogue -----------------------------------------------------
  const int crow0 = m0 + wr * 16 + (lane >> 4) * 8;   // C frag: VGPR r -> M=8*(lane/16)+r
#pragma unroll
  for (int f = 0; f < FR; f++) {
    const int col = n0 + wc * (TN / 2) + f * 16 + (lane & 15);
    if (col >= g.N) continue;
    const float bs = g.bias ? g.bias[col] : 0.f;
#pragma unroll
    for (int r = 0; r < 8; r++) {
      const int row = crow0 + r;
      float v = act_apply(g.alpha * acc[f][r] + bs, g.act);
      const long long idx = co + (long long)row * g.ldc + col;
      if (g.res)  v += g.res[idx];
      if (g.o32)  g.o32[idx] = v;
      if (g.o16)  g.o16[idx] = (_Float16)v;
      if (g.o16t) g.o16t[(long long)col * g.ldct + row] = (_Float16)v;  // transposed side-store
    }
  }
}

// -------------------------- prep / pointwise kernels -----------------------
// out16[n*K+k] = (f16) in[k*N+n]  (tiled transpose+cast of a [K,N] f32 weight)
__launch_bounds__(256)
__global__ void transpose_cast(const float* __restrict__ in, _Float16* __restrict__ out,
                               int K, int N) {
  __shared__ float t[32][33];
  const int bx = blockIdx.x * 32, by = blockIdx.y * 32;
  const int tx = threadIdx.x & 31, ty = threadIdx.x >> 5;
#pragma unroll
  for (int i = 0; i < 32; i += 8) {
    const int k = by + ty + i, n = bx + tx;
    t[ty + i][tx] = (k < K && n < N) ? in[(long long)k * N + n] : 0.f;
  }
  __syncthreads();
#pragma unroll
  for (int i = 0; i < 32; i += 8) {
    const int n = bx + ty + i, k = by + tx;
    if (n < N && k < K) out[(long long)n * K + k] = (_Float16)t[tx][ty + i];
  }
}

__global__ void cast16_k(const float* __restrict__ x, _Float16* __restrict__ y, int n) {
  const int i = blockIdx.x * 256 + threadIdx.x;
  if (i < n) y[i] = (_Float16)x[i];
}
__global__ void zero_f32(float* p, int n) {
  const int i = blockIdx.x * blockDim.x + threadIdx.x;
  if (i < n) p[i] = 0.f;
}

// Row softmax over 1024 cols with fused rel-position gather:
// logits[n,r] = dots[n,r] + S[n, clip(n-r,-512,512)+512]; attn out f16.
__launch_bounds__(256)
__global__ void softmax_rel(const float* __restrict__ dots, const _Float16* __restrict__ S,
                            _Float16* __restrict__ attn) {
  const int n = blockIdx.x, bh = blockIdx.y, tid = threadIdx.x;
  const float*    drow = dots + ((long long)bh * NSEQ + n) * NSEQ;
  const _Float16* srow = S    + ((long long)bh * NSEQ + n) * SREL;
  _Float16*       arow = attn + ((long long)bh * NSEQ + n) * NSEQ;
  float l[4];
  float mx = -1e30f;
#pragma unroll
  for (int j = 0; j < 4; j++) {
    const int r = tid + j * 256;
    int d = n - r;
    d = d > MAXPOS ? MAXPOS : (d < -MAXPOS ? -MAXPOS : d);
    l[j] = drow[r] + (float)srow[d + MAXPOS];
    mx = fmaxf(mx, l[j]);
  }
  __shared__ float red[256];
  red[tid] = mx; __syncthreads();
  for (int s = 128; s > 0; s >>= 1) { if (tid < s) red[tid] = fmaxf(red[tid], red[tid + s]); __syncthreads(); }
  mx = red[0]; __syncthreads();
  float sum = 0.f;
#pragma unroll
  for (int j = 0; j < 4; j++) { l[j] = __expf(l[j] - mx); sum += l[j]; }
  red[tid] = sum; __syncthreads();
  for (int s = 128; s > 0; s >>= 1) { if (tid < s) red[tid] += red[tid + s]; __syncthreads(); }
  const float inv = 1.f / red[0];
#pragma unroll
  for (int j = 0; j < 4; j++) arow[tid + j * 256] = (_Float16)(l[j] * inv);
}

// per-batch sum of (x-bias)^2 (64 chunks/batch, atomic accumulate)
__launch_bounds__(256)
__global__ void bn_reduce(const float* __restrict__ x, const float* __restrict__ bias, float* ms) {
  const int b = blockIdx.y, chunk = blockIdx.x, tid = threadIdx.x;
  const float* xb = x + (long long)b * NSEQ * DIMC + (long long)chunk * (NSEQ / 64) * DIMC;
  float s = 0.f;
  for (int i = tid; i < (NSEQ / 64) * DIMC; i += 256) {
    const float d = xb[i] - bias[i % DIMC];
    s += d * d;
  }
  __shared__ float red[256];
  red[tid] = s; __syncthreads();
  for (int k = 128; k > 0; k >>= 1) { if (tid < k) red[tid] += red[tid + k]; __syncthreads(); }
  if (tid == 0) atomicAdd(&ms[b], red[0]);
}

__global__ void bn_apply(const float* __restrict__ x, const float* __restrict__ w,
                         const float* __restrict__ ms, _Float16* __restrict__ y16) {
  const int i = blockIdx.x * 256 + threadIdx.x;
  if (i >= TOKENS * DIMC) return;
  const int b = i / (NSEQ * DIMC), c = i % DIMC;
  const float inv = rsqrtf(ms[b] * (1.f / (float)(NSEQ * DIMC)));
  y16[i] = (_Float16)(x[i] * inv * __expf(w[c]));
}

// depthwise conv k=31, same-pad 15/15, + bias, + swooshR, f16 out
__launch_bounds__(256)
__global__ void dwconv(const float* __restrict__ y, const float* __restrict__ w,
                       const float* __restrict__ bias, _Float16* __restrict__ out) {
  const int c = blockIdx.x * 256 + threadIdx.x;
  const int n = blockIdx.y, b = blockIdx.z;
  const float* yb = y + (long long)b * NSEQ * INNER2 + c;
  float acc = bias[c];
#pragma unroll
  for (int t = 0; t < KW; t++) {
    const int m = n + t - 15;
    if (m >= 0 && m < NSEQ) acc += w[c * KW + t] * yb[(long long)m * INNER2];
  }
  out[((long long)b * NSEQ + n) * INNER2 + c] = (_Float16)act_apply(acc, 2);
}

__global__ void nla_combine(const _Float16* __restrict__ c16, const _Float16* __restrict__ btA16,
                            const _Float16* __restrict__ a16, _Float16* __restrict__ o16) {
  const int i = blockIdx.x * 256 + threadIdx.x;
  if (i >= TOKENS * D34) return;
  o16[i] = (_Float16)((float)c16[i] * (float)btA16[i] * (float)a16[i]);
}

__global__ void bypass_k(const float* __restrict__ xin, const float* __restrict__ xcur,
                         const float* __restrict__ c, float* __restrict__ xo,
                         _Float16* __restrict__ x16) {
  const int i = blockIdx.x * 256 + threadIdx.x;
  if (i >= TOKENS * DIMC) return;
  const float cc = c[i % DIMC];
  const float v = (1.f - cc) * xin[i] + cc * xcur[i];
  xo[i] = v; x16[i] = (_Float16)v;
}

__global__ void final_k(const float* __restrict__ xin, const float* __restrict__ x,
                        const float* __restrict__ w, const float* __restrict__ ms,
                        const float* __restrict__ c2, float* __restrict__ out) {
  const int i = blockIdx.x * 256 + threadIdx.x;
  if (i >= TOKENS * DIMC) return;
  const int b = i / (NSEQ * DIMC), c = i % DIMC;
  const float inv = rsqrtf(ms[b] * (1.f / (float)(NSEQ * DIMC)));
  const float y = x[i] * inv * __expf(w[c]);
  const float cc = c2[c];
  out[i] = (1.f - cc) * xin[i] + cc * y;
}

// ------------------------------ launcher -----------------------------------
// Input flattening (setup_inputs insertion order, depth-first):
//  0:x | 1-4:ff1{w1,b1,w2,b2} 5-8:ff2 9-12:ff3 | 13:wq 14:wk 15:rel
//  16-23:nla{wa,ba,wb,bb,wc,bc,wl,bl} | 24-26:sa1{wv,wo,bo} 27-29:sa2
//  30-37:conv1{bn_b,bn_w,pw1_w,pw1_b,dw_w,dw_b,pw2_w,pw2_b} 38-45:conv2
//  46:byp1.c 47:byp2.c 48:norm.bias 49:norm.weight
extern "C" void kernel_launch(void* const* d_in, const int* in_sizes, int n_in,
                              void* d_out, int out_size, void* d_ws, size_t ws_size,
                              hipStream_t stream) {
  (void)in_sizes; (void)n_in; (void)out_size; (void)ws_size;
  const float* X = (const float*)d_in[0];
  const float* ffw1[3] = {(const float*)d_in[1], (const float*)d_in[5], (const float*)d_in[9]};
  const float* ffb1[3] = {(const float*)d_in[2], (const float*)d_in[6], (const float*)d_in[10]};
  const float* ffw2[3] = {(const float*)d_in[3], (const float*)d_in[7], (const float*)d_in[11]};
  const float* ffb2[3] = {(const float*)d_in[4], (const float*)d_in[8], (const float*)d_in[12]};
  const float* BA = (const float*)d_in[17];
  const float* BB = (const float*)d_in[19];
  const float* BC = (const float*)d_in[21];
  const float* BL = (const float*)d_in[23];
  const float* sabo[2] = {(const float*)d_in[26], (const float*)d_in[29]};

  // ---- workspace carve-up --------------------------------------------------
  char* W = (char*)d_ws; size_t off = 0;
  auto alloc = [&](size_t bytes) { void* p = W + off; off = (off + bytes + 255) & ~(size_t)255; return p; };
  _Float16* x16    = (_Float16*)alloc((size_t)TOKENS * DIMC * 2);
  _Float16* h16    = (_Float16*)alloc((size_t)TOKENS * FFH * 2);
  float*    xA     = (float*)   alloc((size_t)TOKENS * DIMC * 4);
  float*    xB     = (float*)   alloc((size_t)TOKENS * DIMC * 4);
  _Float16* q16    = (_Float16*)alloc((size_t)TOKENS * DIMC * 2);
  _Float16* k16    = (_Float16*)alloc((size_t)TOKENS * DIMC * 2);
  _Float16* v16t   = (_Float16*)alloc((size_t)DIMC * TOKENS * 2);   // [c][token]
  _Float16* av16   = (_Float16*)alloc((size_t)TOKENS * DIMC * 2);
  _Float16* a16    = (_Float16*)alloc((size_t)TOKENS * D34 * 2);
  _Float16* bt16t  = (_Float16*)alloc((size_t)D34 * TOKENS * 2);    // [c][token]
  _Float16* c16    = (_Float16*)alloc((size_t)TOKENS * D34 * 2);
  _Float16* btA16  = (_Float16*)alloc((size_t)TOKENS * D34 * 2);
  _Float16* nla16  = (_Float16*)alloc((size_t)TOKENS * D34 * 2);
  float*    dots   = (float*)   alloc((size_t)BATCH * HEADS * NSEQ * NSEQ * 4);
  _Float16* S16    = (_Float16*)alloc((size_t)BATCH * HEADS * NSEQ * SREL * 2);
  _Float16* at16   = (_Float16*)alloc((size_t)BATCH * HEADS * NSEQ * NSEQ * 2);
  float*    ycv    = (float*)   alloc((size_t)TOKENS * INNER2 * 4);
  _Float16* yc16   = (_Float16*)alloc((size_t)TOKENS * INNER2 * 2);
  _Float16* y16    = (_Float16*)alloc((size_t)TOKENS * DIMC * 2);
  float*    ms     = (float*)   alloc(64 * 4);
  // f16 [N,K] weights
  _Float16* ffw1T[3]; _Float16* ffw2T[3];
  for (int i = 0; i < 3; i++) {
    ffw1T[i] = (_Float16*)alloc((size_t)FFH * DIMC * 2);
    ffw2T[i] = (_Float16*)alloc((size_t)DIMC * FFH * 2);
  }
  _Float16* wqT  = (_Float16*)alloc((size_t)DIMC * DIMC * 2);
  _Float16* wkT  = (_Float16*)alloc((size_t)DIMC * DIMC * 2);
  _Float16* wvT[2] = {(_Float16*)alloc((size_t)DIMC * DIMC * 2), (_Float16*)alloc((size_t)DIMC * DIMC * 2)};
  _Float16* woT[2] = {(_Float16*)alloc((size_t)DIMC * DIMC * 2), (_Float16*)alloc((size_t)DIMC * DIMC * 2)};
  _Float16* waT  = (_Float16*)alloc((size_t)D34 * DIMC * 2);
  _Float16* wbT  = (_Float16*)alloc((size_t)D34 * DIMC * 2);
  _Float16* wcT  = (_Float16*)alloc((size_t)D34 * DIMC * 2);
  _Float16* wlT  = (_Float16*)alloc((size_t)DIMC * D34 * 2);
  _Float16* rel16  = (_Float16*)alloc((size_t)(2 * MAXPOS + 1) * DIM_HEAD * 2);
  _Float16* pw1c[2] = {(_Float16*)alloc((size_t)INNER2 * DIMC * 2), (_Float16*)alloc((size_t)INNER2 * DIMC * 2)};
  _Float16* pw2c[2] = {(_Float16*)alloc((size_t)DIMC * INNER2 * 2), (_Float16*)alloc((size_t)DIMC * INNER2 * 2)};

  auto TC = [&](const float* w, _Float16* t, int K, int N) {   // [K,N] f32 -> [N,K] f16
    dim3 grid((N + 31) / 32, (K + 31) / 32);
    transpose_cast<<<grid, 256, 0, stream>>>(w, t, K, N);
  };
  auto CAST = [&](const float* w, _Float16* t, int n) {
    cast16_k<<<(n + 255) / 256, 256, 0, stream>>>(w, t, n);
  };

  // ---- weight prep ---------------------------------------------------------
  for (int i = 0; i < 3; i++) {
    TC(ffw1[i], ffw1T[i], DIMC, FFH);
    TC(ffw2[i], ffw2T[i], FFH, DIMC);
  }
  TC((const float*)d_in[13], wqT, DIMC, DIMC);
  TC((const float*)d_in[14], wkT, DIMC, DIMC);
  TC((const float*)d_in[24], wvT[0], DIMC, DIMC); TC((const float*)d_in[27], wvT[1], DIMC, DIMC);
  TC((const float*)d_in[25], woT[0], DIMC, DIMC); TC((const float*)d_in[28], woT[1], DIMC, DIMC);
  TC((const float*)d_in[16], waT, DIMC, D34);
  TC((const float*)d_in[18], wbT, DIMC, D34);
  TC((const float*)d_in[20], wcT, DIMC, D34);
  TC((const float*)d_in[22], wlT, D34, DIMC);
  CAST((const float*)d_in[15], rel16, (2 * MAXPOS + 1) * DIM_HEAD);
  CAST((const float*)d_in[32], pw1c[0], INNER2 * DIMC); CAST((const float*)d_in[40], pw1c[1], INNER2 * DIMC);
  CAST((const float*)d_in[36], pw2c[0], DIMC * INNER2); CAST((const float*)d_in[44], pw2c[1], DIMC * INNER2);

  auto G = [&](const _Float16* A, long long aso, long long asi, int lda,
               const _Float16* B, long long bso, long long bsi, int ldb,
               float* o32, _Float16* o16, _Float16* o16t,
               long long cso, long long csi, int ldc, int ldct,
               const float* bias, const float* res,
               int M, int N, int K, int inner, int batches, float alpha, int act) {
    GemmP g{A, aso, asi, lda, B, bso, bsi, ldb, o32, o16, o16t,
            cso, csi, ldc, ldct, bias, res, M, N, K, inner, alpha, act};
    if (N % 128 == 0) {
      dim3 grid(N / 128, M / 64, batches);
      gemm_wmma<128><<<grid, 256, 0, stream>>>(g);
    } else {
      dim3 grid((N + 63) / 64, M / 64, batches);
      gemm_wmma<64><<<grid, 256, 0, stream>>>(g);
    }
  };
  auto FF = [&](int i, const float* res, float* o32) {
    G(x16, 0, 0, DIMC, ffw1T[i], 0, 0, DIMC, nullptr, h16, nullptr, 0, 0, FFH, 0,
      ffb1[i], nullptr, TOKENS, FFH, DIMC, 1, 1, 1.f, 1);
    G(h16, 0, 0, FFH, ffw2T[i], 0, 0, FFH, o32, x16, nullptr, 0, 0, DIMC, 0,
      ffb2[i], res, TOKENS, DIMC, FFH, 1, 1, 1.f, 1);
  };
  auto SA = [&](int i, const float* res, float* o32) {
    // V projection written directly transposed: v16t[c][token]
    G(x16, 0, 0, DIMC, wvT[i], 0, 0, DIMC, nullptr, nullptr, v16t, 0, 0, DIMC, TOKENS,
      nullptr, nullptr, TOKENS, DIMC, DIMC, 1, 1, 1.f, 0);
    // per-(b,h): attn[1024,1024] @ v[1024,64]  (B rows = v16t rows, cols = tokens of b)
    G(at16, 8LL * NSEQ * NSEQ, (long long)NSEQ * NSEQ, NSEQ,
      v16t, NSEQ, 64LL * TOKENS, TOKENS,
      nullptr, av16, nullptr, (long long)NSEQ * DIMC, DIM_HEAD, DIMC, 0,
      nullptr, nullptr, NSEQ, DIM_HEAD, NSEQ, HEADS, BATCH * HEADS, 1.f, 0);
    G(av16, 0, 0, DIMC, woT[i], 0, 0, DIMC, o32, x16, nullptr, 0, 0, DIMC, 0,
      sabo[i], res, TOKENS, DIMC, DIMC, 1, 1, 1.f, 0);
  };
  auto CONV = [&](int cb, const float* xres, float* o32, int ci) {
    zero_f32<<<1, 64, 0, stream>>>(ms, BATCH);
    bn_reduce<<<dim3(64, BATCH), 256, 0, stream>>>(xres, (const float*)d_in[cb], ms);
    bn_apply<<<(TOKENS * DIMC) / 256, 256, 0, stream>>>(xres, (const float*)d_in[cb + 1], ms, y16);
    G(y16, 0, 0, DIMC, pw1c[ci], 0, 0, DIMC, ycv, nullptr, nullptr, 0, 0, INNER2, 0,
      (const float*)d_in[cb + 3], nullptr, TOKENS, INNER2, DIMC, 1, 1, 1.f, 2);
    dwconv<<<dim3(INNER2 / 256, NSEQ, BATCH), 256, 0, stream>>>(
        ycv, (const float*)d_in[cb + 4], (const float*)d_in[cb + 5], yc16);
    G(yc16, 0, 0, INNER2, pw2c[ci], 0, 0, INNER2, o32, x16, nullptr, 0, 0, DIMC, 0,
      (const float*)d_in[cb + 7], xres, TOKENS, DIMC, INNER2, 1, 1, 1.f, 0);
  };

  const float scale = 0.125f;   // DIM_HEAD^-0.5

  // ---- schedule ------------------------------------------------------------
  cast16_k<<<(TOKENS * DIMC) / 256, 256, 0, stream>>>(X, x16, TOKENS * DIMC);
  FF(0, X, xA);                                                          // x += ff1(x)

  G(x16, 0, 0, DIMC, wqT, 0, 0, DIMC, nullptr, q16, nullptr, 0, 0, DIMC, 0,
    nullptr, nullptr, TOKENS, DIMC, DIMC, 1, 1, 1.f, 0);
  G(x16, 0, 0, DIMC, wkT, 0, 0, DIMC, nullptr, k16, nullptr, 0, 0, DIMC, 0,
    nullptr, nullptr, TOKENS, DIMC, DIMC, 1, 1, 1.f, 0);
  // dots = scale * Q K^T  (B rows = K rows of same (b,h))
  G(q16, (long long)NSEQ * DIMC, DIM_HEAD, DIMC,
    k16, (long long)NSEQ * DIMC, DIM_HEAD, DIMC,
    dots, nullptr, nullptr, 8LL * NSEQ * NSEQ, (long long)NSEQ * NSEQ, NSEQ, 0,
    nullptr, nullptr, NSEQ, NSEQ, DIM_HEAD, HEADS, BATCH * HEADS, scale, 0);
  // S = scale * Q rel^T (shared rel table, N=1025; TDM zero-fills edge rows)
  G(q16, (long long)NSEQ * DIMC, DIM_HEAD, DIMC,
    rel16, 0, 0, DIM_HEAD,
    nullptr, S16, nullptr, 8LL * NSEQ * SREL, (long long)NSEQ * SREL, SREL, 0,
    nullptr, nullptr, NSEQ, 2 * MAXPOS + 1, DIM_HEAD, HEADS, BATCH * HEADS, scale, 0);
  softmax_rel<<<dim3(NSEQ, BATCH * HEADS), 256, 0, stream>>>(dots, S16, at16);

  // ---- NLA -----------------------------------------------------------------
  G(x16, 0, 0, DIMC, waT, 0, 0, DIMC, nullptr, a16, nullptr, 0, 0, D34, 0,
    BA, nullptr, TOKENS, D34, DIMC, 1, 1, 1.f, 0);
  G(x16, 0, 0, DIMC, wbT, 0, 0, DIMC, nullptr, nullptr, bt16t, 0, 0, D34, TOKENS,
    BB, nullptr, TOKENS, D34, DIMC, 1, 1, 1.f, 3);                        // tanh, transposed
  G(x16, 0, 0, DIMC, wcT, 0, 0, DIMC, nullptr, c16, nullptr, 0, 0, D34, 0,
    BC, nullptr, TOKENS, D34, DIMC, 1, 1, 1.f, 0);
  G(at16, 8LL * NSEQ * NSEQ, (long long)NSEQ * NSEQ, NSEQ,
    bt16t, NSEQ, 48LL * TOKENS, TOKENS,
    nullptr, btA16, nullptr, (long long)NSEQ * D34, DH34, D34, 0,
    nullptr, nullptr, NSEQ, DH34, NSEQ, HEADS, BATCH * HEADS, 1.f, 0);
  nla_combine<<<(TOKENS * D34) / 256, 256, 0, stream>>>(c16, btA16, a16, nla16);
  G(nla16, 0, 0, D34, wlT, 0, 0, D34, xB, x16, nullptr, 0, 0, DIMC, 0,
    BL, xA, TOKENS, DIMC, D34, 1, 1, 1.f, 0);                             // cur = xB

  SA(0, xB, xA);                                                          // cur = xA
  CONV(30, xA, xB, 0);                                                    // cur = xB
  FF(1, xB, xA);                                                          // cur = xA
  bypass_k<<<(TOKENS * DIMC) / 256, 256, 0, stream>>>(X, xA, (const float*)d_in[46], xB, x16);
  SA(1, xB, xA);                                                          // cur = xA
  CONV(38, xA, xB, 1);                                                    // cur = xB
  FF(2, xB, xA);                                                          // cur = xA

  zero_f32<<<1, 64, 0, stream>>>(ms, BATCH);
  bn_reduce<<<dim3(64, BATCH), 256, 0, stream>>>(xA, (const float*)d_in[48], ms);
  final_k<<<(TOKENS * DIMC) / 256, 256, 0, stream>>>(
      X, xA, (const float*)d_in[49], ms, (const float*)d_in[47], (float*)d_out);
}
